// Model_1297080124103
// MI455X (gfx1250) — compile-verified, matrix operand order checked by reading, not run on hardware
//
#include <hip/hip_runtime.h>
#include <hip/hip_bf16.h>
#include <math.h>

// ---------------------------------------------------------------------------
// Transformer encoder-decoder forward for gfx1250 (MI455X).
// All GEMMs run through v_wmma_f32_16x16x32_bf16 (f32 accum, bf16 operands
// converted on the fly with v_cvt_pk_bf16_f32 via inline asm). Block tile
// 128x128, 8 waves, K-step 32, double-buffered LDS + register-staged loads.
// ---------------------------------------------------------------------------

typedef __attribute__((ext_vector_type(16))) __bf16 v16bf;
typedef __attribute__((ext_vector_type(8)))  float  v8f;

#define E_DIM   1024
#define FF_DIM  4096
#define VT_DIM  32000
#define B_SZ    2
#define S_LEN   1024
#define NROWS   (B_SZ * S_LEN)   // 2048 flattened (batch, seq) rows
#define EPS_LN  1e-5f

#define TM 128
#define TN 128
#define TK 32
#define A_LDS 40    // shorts per A row: 80B stride -> 16B aligned, conflict-free
#define B_LDS 132   // uints per B pair-row: 528B stride -> 16B aligned

// two f32 -> packed bf16 pair in one VGPR (single VALU op, opaque to opt)
__device__ __forceinline__ unsigned int pk2(float lo, float hi) {
    unsigned int r;
    asm("v_cvt_pk_bf16_f32 %0, %1, %2" : "=v"(r) : "v"(lo), "v"(hi));
    return r;
}

// ---------------------------------------------------------------------------
// GEMM: C[M,N] = act( alpha * A[M,K] @ B + bias ) (+ residual)
//   TRANSB=false: B physical [K,N] row-major (ldb = row stride)
//   TRANSB=true : logical B(k,n) = Bm[n*ldb + k]   (used for Q @ K^T)
// blockIdx.z = batch; sA/sB/sC are per-batch element strides.
// M % 128 == 0, N % 128 == 0, K % 32 == 0 at every call site.
// ---------------------------------------------------------------------------
template<bool TRANSB, bool RELU, bool RES>
__global__ __launch_bounds__(256)
void gemm_wmma(const float* __restrict__ Aall, const float* __restrict__ Ball,
               const float* __restrict__ bias, const float* resall,
               float* Call,
               int M, int N, int K, int ldb,
               long sA, long sB, long sC, float alpha)
{
    // double-buffered staging: A as bf16 shorts, B as pair-packed uints
    __shared__ unsigned short As[2][TM][A_LDS];
    __shared__ unsigned int   Bp[2][TK / 2][B_LDS];

    const int z = blockIdx.z;
    const float* A  = Aall + (long)z * sA;
    const float* Bm = Ball + (long)z * sB;
    const float* res = RES ? (resall + (long)z * sC) : nullptr;
    float* C = Call + (long)z * sC;

    const int tid   = threadIdx.x;
    const int lane  = tid & 31;
    const int wave  = tid >> 5;
    const int l16   = lane & 15;
    const int khalf = (lane >> 4) * 8;     // A short offset half: 0 | 8
    const int kph   = (lane >> 4) * 4;     // B pair-row half:     0 | 4
    const int wr    = (wave >> 2) * 64;    // wave row base within tile
    const int wc    = (wave & 3) * 32;     // wave col base within tile

    const int m0 = blockIdx.y * TM;
    const int n0 = blockIdx.x * TN;

    v8f acc[4][2];
    #pragma unroll
    for (int i = 0; i < 4; ++i)
        #pragma unroll
        for (int j = 0; j < 2; ++j)
            #pragma unroll
            for (int q = 0; q < 8; ++q)
                acc[i][j][q] = 0.0f;

    // B fragment pair-row index per VGPR j
    int kprow[8];
    #pragma unroll
    for (int j = 0; j < 8; ++j)
        kprow[j] = ((j < 4) ? j : (j + 4)) + kph;

    // ---- register staging + loaders ----
    const int a_row = tid >> 1;
    const int a_cb  = (tid & 1) * 16;
    float4 ar[4], br[4];

    auto loadA = [&](int k0) {
        const float4* s =
            reinterpret_cast<const float4*>(&A[(long)(m0 + a_row) * K + k0 + a_cb]);
        ar[0] = s[0]; ar[1] = s[1]; ar[2] = s[2]; ar[3] = s[3];
    };
    auto storeA = [&](int buf) {
        uint4* d = reinterpret_cast<uint4*>(&As[buf][a_row][a_cb]);
        d[0] = make_uint4(pk2(ar[0].x, ar[0].y), pk2(ar[0].z, ar[0].w),
                          pk2(ar[1].x, ar[1].y), pk2(ar[1].z, ar[1].w));
        d[1] = make_uint4(pk2(ar[2].x, ar[2].y), pk2(ar[2].z, ar[2].w),
                          pk2(ar[3].x, ar[3].y), pk2(ar[3].z, ar[3].w));
    };

    auto loadB = [&](int k0) {
        if (!TRANSB) {
            int kp = tid >> 4;              // pair row 0..15
            int cb = (tid & 15) * 8;        // col base
            const float4* s0 =
                reinterpret_cast<const float4*>(&Bm[(long)(k0 + 2 * kp) * ldb + n0 + cb]);
            const float4* s1 =
                reinterpret_cast<const float4*>(&Bm[(long)(k0 + 2 * kp + 1) * ldb + n0 + cb]);
            br[0] = s0[0]; br[1] = s0[1];   // row 2kp   : 8 floats
            br[2] = s1[0]; br[3] = s1[1];   // row 2kp+1 : 8 floats
        } else {
            int nn = tid >> 1;              // column of logical B
            int kb = (tid & 1) * 16;        // k base
            const float4* s =
                reinterpret_cast<const float4*>(&Bm[(long)(n0 + nn) * ldb + k0 + kb]);
            br[0] = s[0]; br[1] = s[1]; br[2] = s[2]; br[3] = s[3];
        }
    };
    auto storeB = [&](int buf) {
        if (!TRANSB) {
            int kp = tid >> 4;
            int cb = (tid & 15) * 8;
            uint4* d = reinterpret_cast<uint4*>(&Bp[buf][kp][cb]);
            d[0] = make_uint4(pk2(br[0].x, br[2].x), pk2(br[0].y, br[2].y),
                              pk2(br[0].z, br[2].z), pk2(br[0].w, br[2].w));
            d[1] = make_uint4(pk2(br[1].x, br[3].x), pk2(br[1].y, br[3].y),
                              pk2(br[1].z, br[3].z), pk2(br[1].w, br[3].w));
        } else {
            int nn  = tid >> 1;
            int kb2 = (tid & 1) * 8;        // pair-row base
            #pragma unroll
            for (int i = 0; i < 4; ++i) {
                Bp[buf][kb2 + 2 * i + 0][nn] = pk2(br[i].x, br[i].y);
                Bp[buf][kb2 + 2 * i + 1][nn] = pk2(br[i].z, br[i].w);
            }
        }
    };

    union FragA { v16bf v; uint4 q[2]; };
    union FragB { v16bf v; unsigned int u[8]; };

    auto compute = [&](int buf) {
        FragA afr[4];
        #pragma unroll
        for (int mt = 0; mt < 4; ++mt) {
            int arow = wr + mt * 16 + l16;
            afr[mt].q[0] = *reinterpret_cast<const uint4*>(&As[buf][arow][khalf]);
            afr[mt].q[1] = *reinterpret_cast<const uint4*>(&As[buf][arow][16 + khalf]);
        }
        #pragma unroll
        for (int nt = 0; nt < 2; ++nt) {
            FragB bfr;
            int bcol = wc + nt * 16 + l16;
            #pragma unroll
            for (int j = 0; j < 8; ++j)
                bfr.u[j] = Bp[buf][kprow[j]][bcol];
            #pragma unroll
            for (int mt = 0; mt < 4; ++mt)
                acc[mt][nt] = __builtin_amdgcn_wmma_f32_16x16x32_bf16(
                    false, afr[mt].v, false, bfr.v,
                    (short)0, acc[mt][nt], false, false);
        }
    };

    // ---- software-pipelined main loop (1 barrier / K-step) ----
    const int nk = K / TK;
    loadA(0); loadB(0);
    storeA(0); storeB(0);
    __syncthreads();
    for (int kt = 0; kt < nk; ++kt) {
        const int cur  = kt & 1;
        const bool more = (kt + 1) < nk;
        if (more) { loadA((kt + 1) * TK); loadB((kt + 1) * TK); }  // issue vmem
        compute(cur);                                              // WMMAs overlap
        if (more) { storeA(cur ^ 1); storeB(cur ^ 1); }            // cvt + LDS store
        __syncthreads();
    }

    // ---- epilogue: alpha, bias, relu, residual ----
    const int rhalf = (lane >> 4) * 8;   // C/D: lanes16-31 hold M=j+8
    #pragma unroll
    for (int mt = 0; mt < 4; ++mt) {
        #pragma unroll
        for (int nt = 0; nt < 2; ++nt) {
            int col = n0 + wc + nt * 16 + l16;
            float bv = bias ? bias[col] : 0.0f;
            #pragma unroll
            for (int j = 0; j < 8; ++j) {
                int row = m0 + wr + mt * 16 + rhalf + j;
                float v = acc[mt][nt][j] * alpha + bv;
                if (RELU) v = fmaxf(v, 0.0f);
                if (RES)  v += res[(long)row * N + col];
                C[(long)row * N + col] = v;
            }
        }
    }
}

// ---------------------------------------------------------------------------
// Row softmax over ncols (=S_LEN), optional causal limit (cols <= q valid).
// ---------------------------------------------------------------------------
__global__ __launch_bounds__(256)
void softmax_rows(float* S, int ncols, int causal)
{
    __shared__ float red[256];
    const int row = blockIdx.x;
    const int q   = row & (S_LEN - 1);
    const int tid = threadIdx.x;
    float* r = S + (long)row * ncols;
    const int lim = causal ? (q + 1) : ncols;

    float mx = -3.402823466e+38f;
    for (int c = tid; c < ncols; c += 256)
        if (c < lim) mx = fmaxf(mx, r[c]);
    red[tid] = mx; __syncthreads();
    for (int s = 128; s > 0; s >>= 1) {
        if (tid < s) red[tid] = fmaxf(red[tid], red[tid + s]);
        __syncthreads();
    }
    mx = red[0]; __syncthreads();

    float sum = 0.0f;
    for (int c = tid; c < ncols; c += 256) {
        float e = (c < lim) ? __expf(r[c] - mx) : 0.0f;
        r[c] = e;
        sum += e;
    }
    red[tid] = sum; __syncthreads();
    for (int s = 128; s > 0; s >>= 1) {
        if (tid < s) red[tid] += red[tid + s];
        __syncthreads();
    }
    float inv = 1.0f / red[0];
    for (int c = tid; c < ncols; c += 256) r[c] *= inv;
}

// ---------------------------------------------------------------------------
// LayerNorm over E_DIM per row (in-place safe: values cached in registers).
// ---------------------------------------------------------------------------
__global__ __launch_bounds__(256)
void layernorm_rows(const float* X, const float* __restrict__ g,
                    const float* __restrict__ b, float* Y)
{
    __shared__ float red[256];
    const int row = blockIdx.x;
    const int tid = threadIdx.x;
    const float* x = X + (long)row * E_DIM;
    float* y = Y + (long)row * E_DIM;

    float v[4];
    float s = 0.0f;
    #pragma unroll
    for (int i = 0; i < 4; ++i) { v[i] = x[tid + i * 256]; s += v[i]; }
    red[tid] = s; __syncthreads();
    for (int st = 128; st > 0; st >>= 1) {
        if (tid < st) red[tid] += red[tid + st];
        __syncthreads();
    }
    float mean = red[0] * (1.0f / E_DIM); __syncthreads();

    float s2 = 0.0f;
    #pragma unroll
    for (int i = 0; i < 4; ++i) { v[i] -= mean; s2 += v[i] * v[i]; }
    red[tid] = s2; __syncthreads();
    for (int st = 128; st > 0; st >>= 1) {
        if (tid < st) red[tid] += red[tid + st];
        __syncthreads();
    }
    float rs = rsqrtf(red[0] * (1.0f / E_DIM) + EPS_LN);

    #pragma unroll
    for (int i = 0; i < 4; ++i) {
        int c = tid + i * 256;
        y[c] = v[i] * rs * g[c] + b[c];
    }
}

// ---------------------------------------------------------------------------
// Token embedding + sinusoidal positional encoding.
// ---------------------------------------------------------------------------
__global__ __launch_bounds__(256)
void embed_pe(const float* __restrict__ emb, const int* __restrict__ seq,
              float* __restrict__ out)
{
    const int row = blockIdx.x;              // 0..NROWS-1
    const int pos = row & (S_LEN - 1);
    const int tid = threadIdx.x;
    const long tok = (long)seq[row];
    const float kf = -0.0089940372f;         // -ln(10000)/1024
    #pragma unroll
    for (int i = 0; i < 4; ++i) {
        int e = tid + i * 256;
        float di  = __expf(kf * (float)(e & ~1));
        float ang = (float)pos * di;
        float pe  = (e & 1) ? __cosf(ang) : __sinf(ang);
        out[(long)row * E_DIM + e] = emb[tok * E_DIM + e] + pe;
    }
}

// ---------------------------------------------------------------------------
// Host orchestration.
// Input flattening: jax tree_flatten (dict keys sorted):
//   dec[0..3]: cross{bk,bo,bq,bv,wk,wo,wq,wv}, ffn{b1,b2,w1,w2},
//              ln1_b,ln1_g,ln2_b,ln2_g,ln3_b,ln3_g,
//              self{bk,bo,bq,bv,wk,wo,wq,wv}                      (26 each)
//   enc[0..3]: ffn{b1,b2,w1,w2}, ln1_b..ln3_g, self{bk..wv}       (18 each)
//   out_b, out_w, src_emb, tgt_emb, src_seq, tgt_seq
// ---------------------------------------------------------------------------
extern "C" void kernel_launch(void* const* d_in, const int* in_sizes, int n_in,
                              void* d_out, int out_size, void* d_ws, size_t ws_size,
                              hipStream_t stream)
{
    (void)in_sizes; (void)n_in; (void)out_size; (void)ws_size;

    auto F = [&](int i) { return (const float*)d_in[i]; };

    const int DEC_STRIDE = 26;
    const int ENC_BASE   = 4 * DEC_STRIDE;            // 104
    const int ENC_STRIDE = 18;
    const int IDX_OUT_B   = ENC_BASE + 4 * ENC_STRIDE; // 176
    const int IDX_OUT_W   = IDX_OUT_B + 1;             // 177
    const int IDX_SRC_EMB = 178;
    const int IDX_TGT_EMB = 179;
    const int IDX_SRC_SEQ = 180;
    const int IDX_TGT_SEQ = 181;

    // workspace layout (floats)
    float* ws = (float*)d_ws;
    const long NT = (long)NROWS * E_DIM;               // 2M floats
    float* x_src = ws;                                 // 2M
    float* x_tgt = x_src + NT;                         // 2M
    float* hbuf  = x_tgt + NT;                         // 2M
    float* qb    = hbuf + NT;                          // 2M
    float* kb    = qb + NT;                            // 2M
    float* vb    = kb + NT;                            // 2M
    float* ctx   = vb + NT;                            // 2M
    float* sc    = ctx + NT;                           // B*S*S = 2M
    float* ffm   = sc + (long)B_SZ * S_LEN * S_LEN;    // 8M

    const dim3 blk(256);
    const float ATT_SCALE = 0.03125f;                  // 1/sqrt(1024)

    auto gemm = [&](const float* A, const float* Bm, const float* bias,
                    const float* res, float* C, int M, int N, int K, int ldb,
                    long sA, long sB, long sC, int batches, float alpha,
                    bool transb, bool relu, bool resf) {
        dim3 grid(N / TN, M / TM, batches);
        if (transb)
            gemm_wmma<true , false, false><<<grid, blk, 0, stream>>>(
                A, Bm, bias, res, C, M, N, K, ldb, sA, sB, sC, alpha);
        else if (relu)
            gemm_wmma<false, true , false><<<grid, blk, 0, stream>>>(
                A, Bm, bias, res, C, M, N, K, ldb, sA, sB, sC, alpha);
        else if (resf)
            gemm_wmma<false, false, true ><<<grid, blk, 0, stream>>>(
                A, Bm, bias, res, C, M, N, K, ldb, sA, sB, sC, alpha);
        else
            gemm_wmma<false, false, false><<<grid, blk, 0, stream>>>(
                A, Bm, bias, res, C, M, N, K, ldb, sA, sB, sC, alpha);
    };

    const long SE  = (long)S_LEN * E_DIM;
    const long SS2 = (long)S_LEN * S_LEN;

    // idx order inside an attn dict: bk,bo,bq,bv,wk,wo,wq,wv
    auto attention = [&](const float* q_in, const float* kv_in, int a0,
                         float* x_inout, bool causal) {
        gemm(q_in , F(a0 + 6), F(a0 + 2), nullptr, qb, NROWS, E_DIM, E_DIM, E_DIM,
             0, 0, 0, 1, 1.0f, false, false, false);                       // Q
        gemm(kv_in, F(a0 + 4), F(a0 + 0), nullptr, kb, NROWS, E_DIM, E_DIM, E_DIM,
             0, 0, 0, 1, 1.0f, false, false, false);                       // K
        gemm(kv_in, F(a0 + 7), F(a0 + 3), nullptr, vb, NROWS, E_DIM, E_DIM, E_DIM,
             0, 0, 0, 1, 1.0f, false, false, false);                       // V
        gemm(qb, kb, nullptr, nullptr, sc, S_LEN, S_LEN, E_DIM, E_DIM,
             SE, SE, SS2, B_SZ, ATT_SCALE, true, false, false);            // Q K^T
        softmax_rows<<<dim3(NROWS), blk, 0, stream>>>(sc, S_LEN, causal ? 1 : 0);
        gemm(sc, vb, nullptr, nullptr, ctx, S_LEN, E_DIM, S_LEN, E_DIM,
             SS2, SE, SE, B_SZ, 1.0f, false, false, false);                // P V
        gemm(ctx, F(a0 + 5), F(a0 + 1), x_inout, x_inout, NROWS, E_DIM, E_DIM, E_DIM,
             0, 0, 0, 1, 1.0f, false, false, true);                        // x += O
    };

    // ---------------- encoder ----------------
    embed_pe<<<dim3(NROWS), blk, 0, stream>>>(F(IDX_SRC_EMB),
                                              (const int*)d_in[IDX_SRC_SEQ], x_src);
    for (int e = 0; e < 4; ++e) {
        int b0 = ENC_BASE + e * ENC_STRIDE;
        // ffn: b1=+0,b2=+1,w1=+2,w2=+3 ; ln1_b=+4,ln1_g=+5,ln2_b=+6,ln2_g=+7,
        // ln3_b=+8,ln3_g=+9 ; self attn base = +10
        layernorm_rows<<<dim3(NROWS), blk, 0, stream>>>(x_src, F(b0+5), F(b0+4), hbuf);
        attention(hbuf, hbuf, b0 + 10, x_src, false);
        layernorm_rows<<<dim3(NROWS), blk, 0, stream>>>(x_src, F(b0+7), F(b0+6), hbuf);
        gemm(hbuf, F(b0+2), F(b0+0), nullptr, ffm, NROWS, FF_DIM, E_DIM, FF_DIM,
             0, 0, 0, 1, 1.0f, false, true, false);                        // relu(W1)
        gemm(ffm, F(b0+3), F(b0+1), x_src, x_src, NROWS, E_DIM, FF_DIM, E_DIM,
             0, 0, 0, 1, 1.0f, false, false, true);                        // x += W2
        layernorm_rows<<<dim3(NROWS), blk, 0, stream>>>(x_src, F(b0+9), F(b0+8), x_src);
    }

    // ---------------- decoder ----------------
    embed_pe<<<dim3(NROWS), blk, 0, stream>>>(F(IDX_TGT_EMB),
                                              (const int*)d_in[IDX_TGT_SEQ], x_tgt);
    for (int d = 0; d < 4; ++d) {
        int b0 = d * DEC_STRIDE;
        // cross attn = +0 ; ffn: b1=+8,b2=+9,w1=+10,w2=+11 ;
        // ln1_b=+12,ln1_g=+13,ln2_b=+14,ln2_g=+15,ln3_b=+16,ln3_g=+17 ; self = +18
        layernorm_rows<<<dim3(NROWS), blk, 0, stream>>>(x_tgt, F(b0+13), F(b0+12), hbuf);
        attention(hbuf, hbuf, b0 + 18, x_tgt, true);                       // causal self
        layernorm_rows<<<dim3(NROWS), blk, 0, stream>>>(x_tgt, F(b0+15), F(b0+14), hbuf);
        attention(hbuf, x_src, b0 + 0, x_tgt, false);                      // cross
        layernorm_rows<<<dim3(NROWS), blk, 0, stream>>>(x_tgt, F(b0+17), F(b0+16), hbuf);
        gemm(hbuf, F(b0+10), F(b0+8), nullptr, ffm, NROWS, FF_DIM, E_DIM, FF_DIM,
             0, 0, 0, 1, 1.0f, false, true, false);
        gemm(ffm, F(b0+11), F(b0+9), x_tgt, x_tgt, NROWS, E_DIM, FF_DIM, E_DIM,
             0, 0, 0, 1, 1.0f, false, false, true);
        // no final norm in decoder layers (faithful to reference)
    }

    // ---------------- logits ----------------
    gemm(x_tgt, F(IDX_OUT_W), F(IDX_OUT_B), nullptr, (float*)d_out,
         NROWS, VT_DIM, E_DIM, VT_DIM, 0, 0, 0, 1, 1.0f, false, false, false);
}
